// Attention_55198919688794
// MI455X (gfx1250) — matrix operand order repaired and checked by reading, not compiled, not run on hardware
//
#include <hip/hip_runtime.h>
#include <hip/hip_bf16.h>

// ---------------------------------------------------------------------------
// Types for CDNA5 WMMA (wave32): v_wmma_f32_16x16x32_f16
// ---------------------------------------------------------------------------
typedef __attribute__((ext_vector_type(16))) _Float16 v16h;
typedef __attribute__((ext_vector_type(8)))  _Float16 half8;
typedef __attribute__((ext_vector_type(4)))  _Float16 half4;
typedef __attribute__((ext_vector_type(8)))  float    v8f;

// Problem constants
#define TOK     16384        // 4 * 4096 tokens
#define DIM     1024
#define INNER   1024
#define QKV_N   3072
#define HEADS   16
#define DHEAD   64
#define CHUNK   64

// Workspace layout (bytes)
#define OFF_XH    0ull                     // x as f16:     16384*1024*2 = 32 MiB
#define OFF_WQKV  33554432ull              // w_qkv f16:    3072*1024*2  =  6 MiB
#define OFF_WOUT  39845888ull              // w_out f16:    1024*1024*2  =  2 MiB
#define OFF_QKV   41943040ull              // qkv f16:      16384*3072*2 = 96 MiB
#define OFF_ATT   142606336ull             // attn out f16: 16384*1024*2 = 32 MiB

// ---------------------------------------------------------------------------
// gfx1250 async global->LDS copy (ASYNCcnt-tracked), with sync fallback.
// Probe: param 1 is '__vector_size__(4*sizeof(int)) int' pointer in AS(1);
// param 2 is the LDS-side pointer (AS(3)); params 3/4 are imm offset / cpol.
// ---------------------------------------------------------------------------
#if defined(__AMDGCN__) && __has_builtin(__builtin_amdgcn_global_load_async_to_lds_b128)
#define HAVE_ASYNC_LDS 1
typedef int gv4i __attribute__((vector_size(16)));
typedef __attribute__((address_space(1))) gv4i as1_v4i;
typedef __attribute__((address_space(3))) gv4i as3_v4i;
#else
#define HAVE_ASYNC_LDS 0
#endif

__device__ __forceinline__ void cp16_lds(_Float16* l, const _Float16* g) {
#if HAVE_ASYNC_LDS
  __builtin_amdgcn_global_load_async_to_lds_b128(
      (as1_v4i*)const_cast<_Float16*>(g), (as3_v4i*)l, 0, 0);
#else
  *(half8*)l = *(const half8*)g;
#endif
}

__device__ __forceinline__ void wait_cp() {
#if HAVE_ASYNC_LDS
#if __has_builtin(__builtin_amdgcn_s_wait_asynccnt)
  __builtin_amdgcn_s_wait_asynccnt(0);
#else
  asm volatile("s_wait_asynccnt 0" ::: "memory");
#endif
#endif
}

// ---------------------------------------------------------------------------
// Fragment loaders (K-major rows in LDS, 16-byte aligned chunks)
// A 16x32 f16: lane&15 = M row, lane>>4 = khalf.
//   elements 0..7  -> K = khalf*8 + 0..7 ; elements 8..15 -> K = khalf*8 + 16..23
// B 32x16 f16: lane&15 = N col, lane>>4 = khalf.
//   elements 0..7  -> K = khalf*8 + 0..7 ; elements 8..15 -> K = khalf*8 + 16..23
// ---------------------------------------------------------------------------
__device__ __forceinline__ v16h load_a_frag(const _Float16* row, int khalf) {
  half8 lo = *(const half8*)(row + khalf * 8);
  half8 hi = *(const half8*)(row + khalf * 8 + 16);
  return __builtin_shufflevector(lo, hi, 0,1,2,3,4,5,6,7,8,9,10,11,12,13,14,15);
}

__device__ __forceinline__ v16h load_b_frag(const _Float16* row, int khalf) {
  half8 lo = *(const half8*)(row + khalf * 8);
  half8 hi = *(const half8*)(row + khalf * 8 + 16);
  return __builtin_shufflevector(lo, hi, 0,1,2,3,4,5,6,7,8,9,10,11,12,13,14,15);
}

__device__ __forceinline__ v8f wmma16(v16h a, v16h b, v8f c) {
  // (neg_a, A, neg_b, B, c_mod, C, reuse_a, reuse_b)
  return __builtin_amdgcn_wmma_f32_16x16x32_f16(false, a, false, b, (short)0, c,
                                                false, false);
}

// ---------------------------------------------------------------------------
// fp32 -> f16 conversion (vectorized x4)
// ---------------------------------------------------------------------------
__global__ __launch_bounds__(256) void cvt_f32_f16(const float* __restrict__ in,
                                                   _Float16* __restrict__ out,
                                                   int n) {
  int i = (blockIdx.x * 256 + threadIdx.x) * 4;
  if (i < n) {
    float4 f = *(const float4*)(in + i);
    half4 h = {(_Float16)f.x, (_Float16)f.y, (_Float16)f.z, (_Float16)f.w};
    *(half4*)(out + i) = h;
  }
}

// ---------------------------------------------------------------------------
// NT GEMM: C[m,n] = sum_k A[m,k] * B[n,k]  (A MxK, B NxK, both f16 row-major)
// Tile 128x128, 256 threads (8 waves); wave w owns rows w*16..w*16+15 and all
// 128 columns -> 8 v8f accumulators. K staged in 64-wide LDS slabs with
// double buffering + async global->LDS copies (16 WMMAs per barrier pair).
// ---------------------------------------------------------------------------
template <bool F32OUT>
__global__ __launch_bounds__(256) void gemm_nt(const _Float16* __restrict__ A,
                                               const _Float16* __restrict__ B,
                                               void* __restrict__ Cv,
                                               const float* __restrict__ bias,
                                               int M, int N, int K) {
  constexpr int BM = 128, BN = 128, BK = 64, STR = 72;  // 144B rows, 16B aligned
  constexpr int ABUF = BM * STR;                        // halfs per A buffer
  __shared__ _Float16 smem[4 * ABUF];                   // A0 A1 B0 B1

  const int tid  = threadIdx.x;
  const int wave = tid >> 5;
  const int lane = tid & 31;
  const int row16 = lane & 15;
  const int khalf = lane >> 4;
  const int m0 = blockIdx.x * BM;
  const int n0 = blockIdx.y * BN;

  const v8f zacc = {0.f, 0.f, 0.f, 0.f, 0.f, 0.f, 0.f, 0.f};
  v8f acc[8];
#pragma unroll
  for (int i = 0; i < 8; ++i) acc[i] = zacc;

  auto stage = [&](int bufi, int k0) {
    _Float16* as_ = smem + bufi * ABUF;            // runtime address math only
    _Float16* bs_ = smem + 2 * ABUF + bufi * ABUF;
#pragma unroll
    for (int c = tid; c < (BM * BK) / 8; c += 256) {
      int r = c >> 3, c8 = (c & 7) * 8;
      cp16_lds(as_ + r * STR + c8, A + (size_t)(m0 + r) * K + k0 + c8);
      cp16_lds(bs_ + r * STR + c8, B + (size_t)(n0 + r) * K + k0 + c8);
    }
  };

  stage(0, 0);
  wait_cp();
  __syncthreads();

  int buf = 0;
  for (int k0 = 0; k0 < K; k0 += BK) {
    if (k0 + BK < K) stage(buf ^ 1, k0 + BK);  // overlaps compute below
    const _Float16* as_ = smem + buf * ABUF;
    const _Float16* bs_ = smem + 2 * ABUF + buf * ABUF;
#pragma unroll
    for (int ks = 0; ks < 2; ++ks) {
      v16h afrag = load_a_frag(as_ + (wave * 16 + row16) * STR + ks * 32, khalf);
#pragma unroll
      for (int nt = 0; nt < 8; ++nt) {
        v16h bfrag = load_b_frag(bs_ + (nt * 16 + row16) * STR + ks * 32, khalf);
        acc[nt] = wmma16(afrag, bfrag, acc[nt]);
      }
    }
    wait_cp();        // next slab resident in LDS
    __syncthreads();  // all waves done reading current slab
    buf ^= 1;
  }

  if (F32OUT) {
    // f32 + bias: store directly (keep f32 precision). Lane col = lane&15,
    // rows = v + 8*khalf.
#pragma unroll
    for (int nt = 0; nt < 8; ++nt) {
#pragma unroll
      for (int v = 0; v < 8; ++v) {
        int gm = m0 + wave * 16 + v + 8 * khalf;
        int gn = n0 + nt * 16 + row16;
        ((float*)Cv)[(size_t)gm * N + gn] = acc[nt][v] + bias[gn];
      }
    }
  } else {
    // f16: repack D tiles through LDS, then coalesced b128 stores.
    constexpr int CSTR = 136;  // 272B rows, 16B aligned
    _Float16* cw = smem + wave * 16 * CSTR;  // 8*16*136 = 17408 <= 4*ABUF
#pragma unroll
    for (int nt = 0; nt < 8; ++nt) {
#pragma unroll
      for (int v = 0; v < 8; ++v)
        cw[(v + 8 * khalf) * CSTR + nt * 16 + row16] = (_Float16)acc[nt][v];
    }
    __syncthreads();
    _Float16* Ch = (_Float16*)Cv;
#pragma unroll
    for (int c = lane; c < 256; c += 32) {  // 16 rows x 16 chunks of 8 halfs
      int r = c >> 4, c8 = (c & 15) * 8;
      *(half8*)(Ch + (size_t)(m0 + wave * 16 + r) * N + n0 + c8) =
          *(const half8*)(cw + r * CSTR + c8);
    }
  }
}

// ---------------------------------------------------------------------------
// Block-local attention. One workgroup (128 threads = 4 waves) per
// (batch, head, chunk). Q,K async-staged row-major; V^T staged via VGPR
// transpose; S = Q K^T via WMMA; f32 row softmax (shfl_xor reductions);
// P staged to LDS (A layout); O = P V via WMMA; coalesced b128 output.
// ---------------------------------------------------------------------------
__global__ __launch_bounds__(128) void attn_kernel(const _Float16* __restrict__ qkv,
                                                   _Float16* __restrict__ o) {
  constexpr int STR = 72;
  __shared__ _Float16 qt[CHUNK * STR];
  __shared__ _Float16 kt[CHUNK * STR];
  __shared__ _Float16 vt[CHUNK * STR];  // transposed: vt[d*STR + j] = V[j,d]
  __shared__ _Float16 pt[CHUNK * STR];

  const int tid  = threadIdx.x;
  const int wave = tid >> 5;
  const int lane = tid & 31;
  const int row16 = lane & 15;
  const int khalf = lane >> 4;

  const int blk = blockIdx.x;         // 4096 = 4 * 16 * 64
  const int b = blk >> 10;
  const int h = (blk >> 6) & 15;
  const int g = blk & 63;
  const size_t tokb = (size_t)b * 4096 + (size_t)g * CHUNK;

  const _Float16* qg = qkv + tokb * QKV_N + h * DHEAD;
  const _Float16* kg = qg + INNER;
  const _Float16* vg = qg + 2 * INNER;

  // Stage tiles: 64x64 halfs each -> 512 8-half chunks.
  for (int c = tid; c < 512; c += 128) {
    int r = c >> 3, c8 = (c & 7) * 8;
    cp16_lds(qt + r * STR + c8, qg + (size_t)r * QKV_N + c8);
    cp16_lds(kt + r * STR + c8, kg + (size_t)r * QKV_N + c8);
    half8 vv = *(const half8*)(vg + (size_t)r * QKV_N + c8);
#pragma unroll
    for (int t = 0; t < 8; ++t) vt[(c8 + t) * STR + r] = vv[t];
  }
  wait_cp();
  __syncthreads();

  const int m0 = wave * 16;
  const v8f zacc = {0.f, 0.f, 0.f, 0.f, 0.f, 0.f, 0.f, 0.f};

  // ---- S = Q K^T; 4 tiles of 16x16 per wave ------------------------------
  v8f s[4];
#pragma unroll
  for (int i = 0; i < 4; ++i) s[i] = zacc;
#pragma unroll
  for (int ks = 0; ks < 2; ++ks) {
    v16h a = load_a_frag(qt + (m0 + row16) * STR + ks * 32, khalf);
#pragma unroll
    for (int nt = 0; nt < 4; ++nt) {
      v16h bf = load_b_frag(kt + (nt * 16 + row16) * STR + ks * 32, khalf);
      s[nt] = wmma16(a, bf, s[nt]);
    }
  }

  // ---- softmax over each 64-wide row, scale = 1/sqrt(64) -----------------
  const float scale = 0.125f;
#pragma unroll
  for (int v = 0; v < 8; ++v) {
    float m = -3.0e38f;
#pragma unroll
    for (int nt = 0; nt < 4; ++nt) m = fmaxf(m, s[nt][v]);
    m = fmaxf(m, __shfl_xor(m, 1));
    m = fmaxf(m, __shfl_xor(m, 2));
    m = fmaxf(m, __shfl_xor(m, 4));
    m = fmaxf(m, __shfl_xor(m, 8));
    m *= scale;
    float p[4];
    float sum = 0.f;
#pragma unroll
    for (int nt = 0; nt < 4; ++nt) {
      p[nt] = __expf(s[nt][v] * scale - m);
      sum += p[nt];
    }
    sum += __shfl_xor(sum, 1);
    sum += __shfl_xor(sum, 2);
    sum += __shfl_xor(sum, 4);
    sum += __shfl_xor(sum, 8);
    float rinv = 1.f / sum;
    int grow = m0 + v + 8 * khalf;
#pragma unroll
    for (int nt = 0; nt < 4; ++nt)
      pt[grow * STR + nt * 16 + row16] = (_Float16)(p[nt] * rinv);
  }
  __syncthreads();

  // ---- O = P V; V^T rows are d, so B frags read contiguous j -------------
  v8f oacc[4];
#pragma unroll
  for (int i = 0; i < 4; ++i) oacc[i] = zacc;
#pragma unroll
  for (int ks = 0; ks < 2; ++ks) {
    v16h a = load_a_frag(pt + (m0 + row16) * STR + ks * 32, khalf);
#pragma unroll
    for (int dt = 0; dt < 4; ++dt) {
      v16h bf = load_b_frag(vt + (dt * 16 + row16) * STR + ks * 32, khalf);
      oacc[dt] = wmma16(a, bf, oacc[dt]);
    }
  }

  // ---- repack O strip into (now free) pt rows, then b128 stores ----------
  _Float16* ow = pt + m0 * STR;  // own 16-row strip, disjoint per wave
#pragma unroll
  for (int dt = 0; dt < 4; ++dt) {
#pragma unroll
    for (int v = 0; v < 8; ++v)
      ow[(v + 8 * khalf) * STR + dt * 16 + row16] = (_Float16)oacc[dt][v];
  }
  __syncthreads();
#pragma unroll
  for (int c = lane; c < 128; c += 32) {  // 16 rows x 8 chunks of 8 halfs
    int r = c >> 3, c8 = (c & 7) * 8;
    *(half8*)(o + (tokb + m0 + r) * INNER + h * DHEAD + c8) =
        *(const half8*)(ow + r * STR + c8);
  }
}

// ---------------------------------------------------------------------------
// Launcher
// ---------------------------------------------------------------------------
extern "C" void kernel_launch(void* const* d_in, const int* in_sizes, int n_in,
                              void* d_out, int out_size, void* d_ws, size_t ws_size,
                              hipStream_t stream) {
  const float* x     = (const float*)d_in[0];
  const float* w_qkv = (const float*)d_in[1];
  const float* w_out = (const float*)d_in[2];
  const float* b_out = (const float*)d_in[3];

  char* ws = (char*)d_ws;
  _Float16* x_h    = (_Float16*)(ws + OFF_XH);
  _Float16* wqkv_h = (_Float16*)(ws + OFF_WQKV);
  _Float16* wout_h = (_Float16*)(ws + OFF_WOUT);
  _Float16* qkv_h  = (_Float16*)(ws + OFF_QKV);
  _Float16* attn_h = (_Float16*)(ws + OFF_ATT);

  // 1) fp32 -> f16 staging
  cvt_f32_f16<<<(TOK * DIM) / 4 / 256, 256, 0, stream>>>(x, x_h, TOK * DIM);
  cvt_f32_f16<<<(QKV_N * DIM) / 4 / 256, 256, 0, stream>>>(w_qkv, wqkv_h, QKV_N * DIM);
  cvt_f32_f16<<<(DIM * INNER) / 4 / 256, 256, 0, stream>>>(w_out, wout_h, DIM * INNER);

  // 2) QKV projection: qkv[n,e] = sum_d x[n,d] * w_qkv[e,d]
  {
    dim3 grid(TOK / 128, QKV_N / 128);
    gemm_nt<false><<<grid, 256, 0, stream>>>(x_h, wqkv_h, qkv_h, nullptr,
                                             TOK, QKV_N, DIM);
  }

  // 3) Block-local attention: 4 batches * 16 heads * 64 chunks
  attn_kernel<<<4 * HEADS * 64, 128, 0, stream>>>(qkv_h, attn_h);

  // 4) Output projection + bias, f32 out
  {
    dim3 grid(TOK / 128, DIM / 128);
    gemm_nt<true><<<grid, 256, 0, stream>>>(attn_h, wout_h, d_out, b_out,
                                            TOK, DIM, INNER);
  }
}
